// FullSelfAttention_16441134809854
// MI455X (gfx1250) — matrix operand order repaired and checked by reading, not compiled, not run on hardware
//
#include <hip/hip_runtime.h>
#include <hip/hip_bf16.h>

// ---------------------------------------------------------------------------
// Causal multi-head self-attention for MI455X (gfx1250), bf16 WMMA pipeline.
//   B=4, N=2048, D=1024, H=16, dh=64  (all dims compile-time constants)
// Stages:
//   1) fp32 -> bf16 convert (x), fp32 -> bf16 transpose (Wq/Wk/Wv/Wo)
//   2) QKV GEMMs (bf16 WMMA, fp32 accum, async-LDS double-buffered staging)
//   3) flash attention: S=QK^T wmma, online softmax (exp2), PV wmma
//   4) output projection GEMM -> fp32 d_out (+bias)
// ---------------------------------------------------------------------------

typedef __attribute__((ext_vector_type(16))) __bf16 v16bf;
typedef __attribute__((ext_vector_type(8)))  float  v8f;
typedef int v4i __attribute__((vector_size(16)));   // matches builtin param type

union Frag {
    v16bf v;
    uint4 q[2];
};

// CDNA5 async global->LDS staging (ASYNCcnt-tracked), if the toolchain has it.
// Probe-learned signature: (v4i addrspace(1)* src, v4i addrspace(3)* dst,
//                           imm int offset, imm int cpol)
#if defined(__AMDGCN__) && \
    __has_builtin(__builtin_amdgcn_global_load_async_to_lds_b128) && \
    __has_builtin(__builtin_amdgcn_s_wait_asynccnt)
#define USE_ASYNC_LDS 1
#define GPTR(p) ((__attribute__((address_space(1))) v4i*)(p))
#define SPTR(p) ((__attribute__((address_space(3))) v4i*)(p))
#else
#define USE_ASYNC_LDS 0
#endif

#if defined(__AMDGCN__) && __has_builtin(__builtin_amdgcn_rcpf)
#define FAST_RCP(x) __builtin_amdgcn_rcpf(x)
#else
#define FAST_RCP(x) (1.0f / (x))
#endif

__device__ __forceinline__ unsigned short f2bf(float f) {
    unsigned int u = __float_as_uint(f);
    unsigned int r = ((u >> 16) & 1u) + 0x7FFFu;   // round-to-nearest-even
    return (unsigned short)((u + r) >> 16);
}

__device__ __forceinline__ v8f wmma_bf16(const Frag& a, const Frag& b, v8f c) {
    return __builtin_amdgcn_wmma_f32_16x16x32_bf16(
        false, a.v, false, b.v, (short)0, c, false, false);
}

__device__ __forceinline__ v8f v8f_zero() {
    return (v8f){0.f, 0.f, 0.f, 0.f, 0.f, 0.f, 0.f, 0.f};
}

// problem constants
#define C_B    4
#define C_NSEQ 2048
#define C_D    1024
#define C_H    16
#define C_DH   64

// ---------------------------------------------------------------------------
// Stage 1a: elementwise fp32 -> bf16 (n multiple of 4)
// ---------------------------------------------------------------------------
__global__ __launch_bounds__(256) void k_f32_to_bf16(
    const float* __restrict__ in, unsigned short* __restrict__ out, long n)
{
    long i = ((long)blockIdx.x * blockDim.x + threadIdx.x) * 4;
    if (i + 3 >= n) return;
    float4 f = *(const float4*)(in + i);
    uint2 p;
    p.x = (unsigned)f2bf(f.x) | ((unsigned)f2bf(f.y) << 16);
    p.y = (unsigned)f2bf(f.z) | ((unsigned)f2bf(f.w) << 16);
    *(uint2*)(out + i) = p;
}

// ---------------------------------------------------------------------------
// Stage 1b: fp32 [R x C] -> bf16 transposed [C x R]
// ---------------------------------------------------------------------------
__global__ __launch_bounds__(256) void k_transpose_bf16(
    const float* __restrict__ in, unsigned short* __restrict__ out, int R, int C)
{
    long idx = (long)blockIdx.x * blockDim.x + threadIdx.x;
    if (idx >= (long)R * C) return;
    int r = (int)(idx / C), c = (int)(idx % C);
    out[(long)c * R + r] = f2bf(in[idx]);
}

// ---------------------------------------------------------------------------
// Stage 2/4: GEMM  C[M x 1024] = A[M x 1024] * Bt[1024 x 1024]^T (bf16, f32 acc)
//   128x128 tile, 256 threads = 8 waves, each wave 32x64 (2x4 WMMA subtiles).
//   LDS stride 40 bf16 (pad) against bank conflicts; async double-buffered
//   staging when GLOBAL_LOAD_ASYNC_TO_LDS is available.
// MODE 0: bf16 head-split  out[((b*H+h)*N+n)*64+d]           (Q, K)
// MODE 1: bf16 head-split transposed out[((b*H+h)*64+d)*N+n] (V^T, b128 store)
// MODE 2: fp32 flat out[m*1024+o] = acc + bias[o]            (final proj)
// ---------------------------------------------------------------------------
#define LSTR 40

template<int MODE>
__global__ __launch_bounds__(256) void k_gemm_bf16(
    const unsigned short* __restrict__ A,
    const unsigned short* __restrict__ Bt,
    void* __restrict__ Out,
    const float* __restrict__ bias)
{
    constexpr int Kdim = C_D;
    constexpr int NS   = Kdim / 32;

    const int tid  = threadIdx.x;
    const int lane = tid & 31;
    const int wid  = tid >> 5;
    const int wm   = wid & 3;    // 4 wave-rows of 32
    const int wn   = wid >> 2;   // 2 wave-cols of 64
    const long m0  = (long)blockIdx.y * 128;
    const long n0  = (long)blockIdx.x * 128;

    const int lr = lane & 15;
    const int hi = lane >> 4;
    const int kA = hi * 8;    // A-frag K half offset (bf16 elems)
    const int kB = hi * 8;    // B-frag K half offset (bf16 elems)

    v8f acc[2][4];
    for (int i = 0; i < 2; ++i)
        for (int j = 0; j < 4; ++j)
            acc[i][j] = v8f_zero();

#if USE_ASYNC_LDS
    __shared__ unsigned short lA[2][128 * LSTR];
    __shared__ unsigned short lB[2][128 * LSTR];

    auto issue = [&](int buf, int k0) {
        #pragma unroll
        for (int i = 0; i < 2; ++i) {
            int cid = tid + i * 256;       // 0..511
            int row = cid >> 2;            // 0..127
            int c8  = (cid & 3) * 8;       // bf16 col offset in 32-wide slab
            __builtin_amdgcn_global_load_async_to_lds_b128(
                GPTR(&A[(m0 + row) * Kdim + k0 + c8]),
                SPTR(&lA[buf][row * LSTR + c8]), 0, 0);
            __builtin_amdgcn_global_load_async_to_lds_b128(
                GPTR(&Bt[(n0 + row) * Kdim + k0 + c8]),
                SPTR(&lB[buf][row * LSTR + c8]), 0, 0);
        }
    };

    issue(0, 0);
    for (int s = 0; s < NS; ++s) {
        const int cur = s & 1;
        __builtin_amdgcn_s_wait_asynccnt(0);   // this wave's async writes landed
        __syncthreads();                       // everyone's landed, prev readers done
        if (s + 1 < NS) issue(1 - cur, (s + 1) * 32);

        Frag fa[2], fb[4];
        #pragma unroll
        for (int i = 0; i < 2; ++i) {
            const unsigned short* p = &lA[cur][(wm * 32 + i * 16 + lr) * LSTR + kA];
            fa[i].q[0] = *(const uint4*)p;
            fa[i].q[1] = *(const uint4*)(p + 16);
        }
        #pragma unroll
        for (int j = 0; j < 4; ++j) {
            const unsigned short* p = &lB[cur][(wn * 64 + j * 16 + lr) * LSTR + kB];
            fb[j].q[0] = *(const uint4*)p;
            fb[j].q[1] = *(const uint4*)(p + 16);
        }
        #pragma unroll
        for (int i = 0; i < 2; ++i)
            #pragma unroll
            for (int j = 0; j < 4; ++j)
                acc[i][j] = wmma_bf16(fa[i], fb[j], acc[i][j]);
    }
#else
    __shared__ unsigned short lA[128 * LSTR];
    __shared__ unsigned short lB[128 * LSTR];

    for (int s = 0; s < NS; ++s) {
        const int k0 = s * 32;
        #pragma unroll
        for (int i = 0; i < 2; ++i) {
            int cid = tid + i * 256;
            int row = cid >> 2;
            int c8  = (cid & 3) * 8;
            *(uint4*)&lA[row * LSTR + c8] = *(const uint4*)&A[(m0 + row) * Kdim + k0 + c8];
            *(uint4*)&lB[row * LSTR + c8] = *(const uint4*)&Bt[(n0 + row) * Kdim + k0 + c8];
        }
        __syncthreads();
        Frag fa[2], fb[4];
        #pragma unroll
        for (int i = 0; i < 2; ++i) {
            const unsigned short* p = &lA[(wm * 32 + i * 16 + lr) * LSTR + kA];
            fa[i].q[0] = *(const uint4*)p;
            fa[i].q[1] = *(const uint4*)(p + 16);
        }
        #pragma unroll
        for (int j = 0; j < 4; ++j) {
            const unsigned short* p = &lB[(wn * 64 + j * 16 + lr) * LSTR + kB];
            fb[j].q[0] = *(const uint4*)p;
            fb[j].q[1] = *(const uint4*)(p + 16);
        }
        #pragma unroll
        for (int i = 0; i < 2; ++i)
            #pragma unroll
            for (int j = 0; j < 4; ++j)
                acc[i][j] = wmma_bf16(fa[i], fb[j], acc[i][j]);
        __syncthreads();
    }
#endif

    // ---- store epilogue (all index math is shifts/masks) ----
    #pragma unroll
    for (int i = 0; i < 2; ++i)
        #pragma unroll
        for (int j = 0; j < 4; ++j) {
            long mrow0 = m0 + wm * 32 + i * 16 + 8 * hi;   // rows mrow0..mrow0+7
            long o     = n0 + wn * 64 + j * 16 + lr;       // output column
            if (MODE == 1) {
                // V^T: rows map to contiguous n -> one b128 store of 8 bf16
                int b = (int)(mrow0 >> 11), n = (int)(mrow0 & 2047);
                int h = (int)(o >> 6),      d = (int)(o & 63);
                uint4 pk;
                pk.x = (unsigned)f2bf(acc[i][j][0]) | ((unsigned)f2bf(acc[i][j][1]) << 16);
                pk.y = (unsigned)f2bf(acc[i][j][2]) | ((unsigned)f2bf(acc[i][j][3]) << 16);
                pk.z = (unsigned)f2bf(acc[i][j][4]) | ((unsigned)f2bf(acc[i][j][5]) << 16);
                pk.w = (unsigned)f2bf(acc[i][j][6]) | ((unsigned)f2bf(acc[i][j][7]) << 16);
                long idx = (((long)b * C_H + h) * C_DH + d) * (long)C_NSEQ + n;
                *(uint4*)&((unsigned short*)Out)[idx] = pk;
            } else {
                for (int r = 0; r < 8; ++r) {
                    long m = mrow0 + r;
                    float val = acc[i][j][r];
                    if (MODE == 2) {
                        ((float*)Out)[m * C_D + o] = val + bias[(int)o];
                    } else {
                        int b = (int)(m >> 11), n = (int)(m & 2047);
                        int h = (int)(o >> 6),  d = (int)(o & 63);
                        long idx = (((long)b * C_H + h) * C_NSEQ + n) * C_DH + d;
                        ((unsigned short*)Out)[idx] = f2bf(val);
                    }
                }
            }
        }
}

// ---------------------------------------------------------------------------
// Stage 3: causal flash attention, one wave = 16 query rows.
//   Q,K: [B,H,N,64] bf16; Vt: [B,H,64,N] bf16; Out: [B,N,H*64] bf16
//   Softmax in exp2 domain: scale2 = dh^-0.5 * log2(e)
// ---------------------------------------------------------------------------
__global__ __launch_bounds__(256) void k_flash_attn(
    const unsigned short* __restrict__ Q,
    const unsigned short* __restrict__ K,
    const unsigned short* __restrict__ Vt,
    unsigned short* __restrict__ Oout)
{
    __shared__ unsigned short lP[8][16 * 72];   // per-wave P staging, pad to 72

    constexpr int DH = C_DH, Nseq = C_NSEQ;
    const float scale2 = 0.125f * 1.44269504089f;   // dh^-1/2 * log2(e)

    const int tid  = threadIdx.x;
    const int lane = tid & 31;
    const int wid  = tid >> 5;
    const int bh   = blockIdx.y;
    const int b    = bh >> 4, h = bh & 15;
    const int m0   = blockIdx.x * 128 + wid * 16;   // wave's first query row

    const long base = (long)bh * Nseq * DH;
    const unsigned short* qp = Q  + base;
    const unsigned short* kp = K  + base;
    const unsigned short* vp = Vt + base;           // [64][Nseq]

    const int lr = lane & 15;
    const int hi = lane >> 4;
    const int kA = hi * 8;
    const int kB = hi * 8;

    // Q fragments (K-dim = dh = 64 -> 2 chunks of 32), kept in registers
    Frag fq[2];
    #pragma unroll
    for (int c = 0; c < 2; ++c) {
        const unsigned short* p = qp + (long)(m0 + lr) * DH + c * 32 + kA;
        fq[c].q[0] = *(const uint4*)p;
        fq[c].q[1] = *(const uint4*)(p + 16);
    }

    v8f o_acc[4];
    for (int j = 0; j < 4; ++j) o_acc[j] = v8f_zero();
    float mrow[8], lrow[8];
    for (int r = 0; r < 8; ++r) { mrow[r] = -3.0e38f; lrow[r] = 0.f; }

    unsigned short* pb = &lP[wid][0];
    const int kend = m0 + 15;   // last key this wave can see (causal)

    for (int kb = 0; kb <= kend; kb += 64) {
        // hint next key block into cache while this one computes
        if (kb + 64 <= kend) {
            __builtin_prefetch(kp + (long)(kb + 64 + lane) * DH, 0, 1);
            __builtin_prefetch(vp + (long)lane * Nseq + kb + 64, 0, 1);
        }

        // ---- S = Q K^T for 4 key n-tiles of 16 ----
        v8f s[4];
        #pragma unroll
        for (int j = 0; j < 4; ++j) {
            v8f sj = v8f_zero();
            #pragma unroll
            for (int c = 0; c < 2; ++c) {
                Frag fk;   // B operand: column = K-row (key kb+j*16+lr)
                const unsigned short* p = kp + (long)(kb + j * 16 + lr) * DH + c * 32 + kB;
                fk.q[0] = *(const uint4*)p;
                fk.q[1] = *(const uint4*)(p + 16);
                sj = wmma_bf16(fq[c], fk, sj);
            }
            s[j] = sj;
        }

        // ---- scale (log2 domain) + causal mask (wave-uniform predicate) ----
        const bool needMask = (kb + 63) > m0;
        for (int j = 0; j < 4; ++j)
            for (int r = 0; r < 8; ++r) {
                float x = s[j][r] * scale2;
                if (needMask) {
                    int qi = m0 + r + 8 * hi;
                    int ki = kb + j * 16 + lr;
                    x = (ki <= qi) ? x : -3.0e38f;
                }
                s[j][r] = x;
            }

        // ---- online softmax: row max across 16-lane half, rescale ----
        float corr[8];
        for (int r = 0; r < 8; ++r) {
            float mx = fmaxf(fmaxf(s[0][r], s[1][r]), fmaxf(s[2][r], s[3][r]));
            for (int off = 1; off < 16; off <<= 1)
                mx = fmaxf(mx, __shfl_xor(mx, off, 32));
            float mnew = fmaxf(mrow[r], mx);
            corr[r] = exp2f(mrow[r] - mnew);
            mrow[r] = mnew;
        }

        // ---- P = exp2(S - m), stage to LDS (A-layout rows), update l ----
        for (int r = 0; r < 8; ++r) {
            float acc_l = 0.f;
            int row = r + 8 * hi;
            for (int j = 0; j < 4; ++j) {
                float p = exp2f(s[j][r] - mrow[r]);
                acc_l += p;
                pb[row * 72 + j * 16 + lr] = f2bf(p);
            }
            for (int off = 1; off < 16; off <<= 1)
                acc_l += __shfl_xor(acc_l, off, 32);
            lrow[r] = lrow[r] * corr[r] + acc_l;
        }
        for (int j = 0; j < 4; ++j)
            for (int r = 0; r < 8; ++r)
                o_acc[j][r] *= corr[r];

        // ---- O += P * V  (A = P from LDS, B = V^T rows, contiguous) ----
        #pragma unroll
        for (int j = 0; j < 4; ++j) {          // dh n-tiles
            #pragma unroll
            for (int c = 0; c < 2; ++c) {      // key chunks of 32
                Frag fp, fv;
                const unsigned short* pp = pb + lr * 72 + c * 32 + kA;
                fp.q[0] = *(const uint4*)pp;
                fp.q[1] = *(const uint4*)(pp + 16);
                const unsigned short* pv = vp + (long)(j * 16 + lr) * Nseq + kb + c * 32 + kB;
                fv.q[0] = *(const uint4*)pv;
                fv.q[1] = *(const uint4*)(pv + 16);
                o_acc[j] = wmma_bf16(fp, fv, o_acc[j]);
            }
        }
    }

    // ---- finalize: O * (1/l) -> attn-out bf16 [B,N,H*64] ----
    float rinv[8];
    for (int r = 0; r < 8; ++r) rinv[r] = FAST_RCP(lrow[r]);
    for (int j = 0; j < 4; ++j)
        for (int r = 0; r < 8; ++r) {
            int qi = m0 + r + 8 * hi;
            float val = o_acc[j][r] * rinv[r];
            long idx = ((long)b * Nseq + qi) * (long)C_D + h * DH + j * 16 + lr;
            Oout[idx] = f2bf(val);
        }
}

// ---------------------------------------------------------------------------
// Launch
// ---------------------------------------------------------------------------
extern "C" void kernel_launch(void* const* d_in, const int* in_sizes, int n_in,
                              void* d_out, int out_size, void* d_ws, size_t ws_size,
                              hipStream_t stream)
{
    (void)in_sizes; (void)n_in; (void)out_size; (void)ws_size;
    const float* x  = (const float*)d_in[0];
    const float* Wq = (const float*)d_in[1];
    const float* Wk = (const float*)d_in[2];
    const float* Wv = (const float*)d_in[3];
    const float* Wo = (const float*)d_in[4];
    const float* bo = (const float*)d_in[5];
    float* out = (float*)d_out;

    const int D = C_D, Nseq = C_NSEQ, B = C_B, H = C_H;
    const long MT = (long)B * Nseq;   // 8192 rows

    // workspace layout (bytes)
    char* ws = (char*)d_ws;
    const long MB = 1l << 20;
    unsigned short* xb   = (unsigned short*)(ws);             // 16 MB
    unsigned short* WqT  = (unsigned short*)(ws + 16 * MB);   //  2 MB
    unsigned short* WkT  = (unsigned short*)(ws + 18 * MB);
    unsigned short* WvT  = (unsigned short*)(ws + 20 * MB);
    unsigned short* WoT  = (unsigned short*)(ws + 22 * MB);
    unsigned short* qbuf = (unsigned short*)(ws + 24 * MB);   // 16 MB [B,H,N,64]
    unsigned short* kbuf = (unsigned short*)(ws + 40 * MB);   // 16 MB [B,H,N,64]
    unsigned short* vtbf = (unsigned short*)(ws + 56 * MB);   // 16 MB [B,H,64,N]
    unsigned short* attb = (unsigned short*)(ws + 72 * MB);   // 16 MB [B,N,1024]

    // 1) conversions
    {
        long n = MT * D;
        k_f32_to_bf16<<<dim3((unsigned)((n / 4 + 255) / 256)), 256, 0, stream>>>(x, xb, n);
        unsigned tb = (unsigned)(((long)D * D + 255) / 256);
        k_transpose_bf16<<<tb, 256, 0, stream>>>(Wq, WqT, D, D);
        k_transpose_bf16<<<tb, 256, 0, stream>>>(Wk, WkT, D, D);
        k_transpose_bf16<<<tb, 256, 0, stream>>>(Wv, WvT, D, D);
        k_transpose_bf16<<<tb, 256, 0, stream>>>(Wo, WoT, D, D);
    }

    // 2) QKV projections
    dim3 gg(D / 128, (unsigned)(MT / 128));   // (8, 64)
    k_gemm_bf16<0><<<gg, 256, 0, stream>>>(xb, WqT, qbuf, nullptr);
    k_gemm_bf16<0><<<gg, 256, 0, stream>>>(xb, WkT, kbuf, nullptr);
    k_gemm_bf16<1><<<gg, 256, 0, stream>>>(xb, WvT, vtbf, nullptr);

    // 3) causal flash attention
    dim3 ga(Nseq / 128, B * H);               // (16, 64)
    k_flash_attn<<<ga, 256, 0, stream>>>(qbuf, kbuf, vtbf, attb);

    // 4) output projection (+bias) -> fp32 d_out
    k_gemm_bf16<2><<<gg, 256, 0, stream>>>(attb, WoT, out, bo);
}
